// LocalizerBranch_28028956574145
// MI455X (gfx1250) — matrix operand order repaired and checked by reading, not compiled, not run on hardware
//
#include <hip/hip_runtime.h>
#include <hip/hip_bf16.h>

// Problem dims (fixed by the reference)
#define BDIM 64
#define SDIM 8
#define DDIM 256
#define FDIM 256

typedef __attribute__((ext_vector_type(8)))  _Float16 v8h;
typedef __attribute__((ext_vector_type(16))) _Float16 v16h;
typedef __attribute__((ext_vector_type(8)))  float    v8f;

// ---------------------------------------------------------------------------
// Kernel 1: per-(b,s) key feature: gathered slice if index valid, else mean
// over D. Grid = B*S blocks, 256 threads (one per feature f).
// Valid path: 1 coalesced KB per block. Invalid path: 256 coalesced rows.
// ---------------------------------------------------------------------------
__global__ __launch_bounds__(FDIM) void lb_gather_or_mean(
    const float* __restrict__ features,      // [B,S,D,F]
    const long long* __restrict__ key_slices,// [B,S] int64
    float* __restrict__ keyfeat)             // [B*S, F] (workspace)
{
    const int bs = blockIdx.x;               // 0 .. B*S-1
    const int f  = threadIdx.x;              // 0 .. F-1
    const long long ks = key_slices[bs];
    const float* base = features + (size_t)bs * DDIM * FDIM;

    float v;
    if (ks >= 0 && ks < DDIM) {
        v = base[(int)ks * FDIM + f];
    } else {
        float acc = 0.0f;
#pragma unroll 8
        for (int d = 0; d < DDIM; ++d)
            acc += base[d * FDIM + f];
        v = acc * (1.0f / (float)DDIM);
    }
    keyfeat[bs * FDIM + f] = v;
}

// ---------------------------------------------------------------------------
// Kernel 2: local[b,f] = mean over S of keyfeat, converted to f16 for WMMA A.
// ---------------------------------------------------------------------------
__global__ __launch_bounds__(FDIM) void lb_mean_s_to_f16(
    const float* __restrict__ keyfeat,       // [B*S, F]
    _Float16* __restrict__ localh)           // [B, F] f16
{
    const int b = blockIdx.x;
    const int f = threadIdx.x;
    float acc = 0.0f;
#pragma unroll
    for (int s = 0; s < SDIM; ++s)
        acc += keyfeat[(b * SDIM + s) * FDIM + f];
    localh[b * FDIM + f] = (_Float16)(acc * (1.0f / (float)SDIM));
}

// ---------------------------------------------------------------------------
// Kernel 3: W (f32 [F,F], torch Linear layout W[out,in]) -> f16. Row-major W
// is exactly the B-matrix feed: column n of B = W^T is row n of W.
// ---------------------------------------------------------------------------
__global__ __launch_bounds__(256) void lb_w_to_f16(
    const float* __restrict__ W, _Float16* __restrict__ Wh)
{
    const int i = blockIdx.x * 256 + threadIdx.x;   // 256*256 elements
    Wh[i] = (_Float16)W[i];
}

// ---------------------------------------------------------------------------
// Kernel 4: out = relu(local @ W^T + b) via V_WMMA_F32_16X16X32_F16.
// M=64, N=256, K=256 -> 4x16 = 64 tiles of 16x16, one wave32 per tile,
// 8 WMMA instructions per wave (K/32). 8 blocks x 8 waves.
// ---------------------------------------------------------------------------
__global__ __launch_bounds__(256) void lb_wmma_linear_relu(
    const _Float16* __restrict__ Ah,   // [64,256]  local (row-major)
    const _Float16* __restrict__ Bh,   // [256,256] W (row-major) == B^T feed
    const float* __restrict__ bias,    // [256]
    float* __restrict__ out)           // [64,256]
{
    const int wave = (int)(blockIdx.x * blockDim.x + threadIdx.x) >> 5; // 0..63
    const int lane = (int)threadIdx.x & 31;

    const int tm = wave & 3;    // M tile: 0..3
    const int tn = wave >> 2;   // N tile: 0..15

    const int m      = tm * 16 + (lane & 15);   // A row for this lane
    const int n      = tn * 16 + (lane & 15);   // B column / D column
    const int khalf  = (lane >> 4) * 8;         // unused; A and B both use kbase
    const int kbase  = (lane >> 4) * 16;        // B: K sub-offset per lane half

    v8f c = {};
#pragma unroll
    for (int kb = 0; kb < FDIM; kb += 32) {
        // A 16x32 f16: elems 0..15 = K[kbase..kbase+15]
        v8h alo = *(const v8h*)(Ah + m * FDIM + kb + kbase);
        v8h ahi = *(const v8h*)(Ah + m * FDIM + kb + kbase + 8);
        v16h a = __builtin_shufflevector(alo, ahi,
                 0, 1, 2, 3, 4, 5, 6, 7, 8, 9, 10, 11, 12, 13, 14, 15);
        // B 32x16 f16: lane = col n, elems 0..15 = K[kbase..kbase+15]
        v16h bm = *(const v16h*)(Bh + n * FDIM + kb + kbase);
        // D = A x B + C  (f32 accumulate)
        c = __builtin_amdgcn_wmma_f32_16x16x32_f16(
                /*neg_a=*/false, a, /*neg_b=*/false, bm,
                /*c_mod=*/(short)0, c, /*reuse_a=*/false, /*reuse_b=*/false);
    }

    // C/D layout: VGPR r -> M = (lane>>4)*8 + r, N = lane&15
    const float bn   = bias[n];
    const int   mrow = tm * 16 + (lane >> 4) * 8;
#pragma unroll
    for (int r = 0; r < 8; ++r) {
        float v = c[r] + bn;
        out[(mrow + r) * FDIM + n] = v > 0.0f ? v : 0.0f;
    }
}

// ---------------------------------------------------------------------------
// Launch
// ---------------------------------------------------------------------------
extern "C" void kernel_launch(void* const* d_in, const int* in_sizes, int n_in,
                              void* d_out, int out_size, void* d_ws, size_t ws_size,
                              hipStream_t stream) {
    (void)in_sizes; (void)n_in; (void)out_size; (void)ws_size;

    const float*     features   = (const float*)d_in[0];       // [64,8,256,256]
    const long long* key_slices = (const long long*)d_in[1];   // [64,8] int64
    const float*     W          = (const float*)d_in[2];       // [256,256]
    const float*     bias       = (const float*)d_in[3];       // [256]
    float*           out        = (float*)d_out;               // [64,256]

    // Workspace carve-up (all offsets 128B-aligned)
    char* ws = (char*)d_ws;
    float*    keyfeat = (float*)ws;                                  // 512 KB
    _Float16* localh  = (_Float16*)(ws + (size_t)BDIM * SDIM * FDIM * 4); // 32 KB
    _Float16* Wh      = (_Float16*)(ws + (size_t)BDIM * SDIM * FDIM * 4
                                       + (size_t)BDIM * FDIM * 2);   // 128 KB

    lb_gather_or_mean<<<BDIM * SDIM, FDIM, 0, stream>>>(features, key_slices, keyfeat);
    lb_mean_s_to_f16 <<<BDIM, FDIM, 0, stream>>>(keyfeat, localh);
    lb_w_to_f16      <<<(FDIM * FDIM) / 256, 256, 0, stream>>>(W, Wh);
    lb_wmma_linear_relu<<<8, 256, 0, stream>>>(localh, Wh, bias, out);
}
